// MogrifierRNN_28363964023664
// MI455X (gfx1250) — matrix operand order repaired and checked by reading, not compiled, not run on hardware
//
#include <hip/hip_runtime.h>
#include <hip/hip_bf16.h>

// ---------------------------------------------------------------------------
// Mogrifier-LSTM, persistent-kernel implementation for MI455X (gfx1250).
//   S=512 steps, B=128, M=N=512, K(rank)=256, NQ=2.
// bf16 WMMA (v_wmma_f32_16x16x32_bf16) for all GEMMs, fp32 state.
// 32 persistent workgroups (256 wave32s); each wave owns a 16x16 output tile.
// Activation A-tiles staged in LDS (shared by the 8 waves of a block);
// weights stream from L2 (global_load_b128 + global_prefetch_b8).
// Grid-wide spin barrier between GEMM phases.
// ---------------------------------------------------------------------------

typedef __bf16 bf16;
typedef __attribute__((ext_vector_type(16))) __bf16 v16bf;
typedef __attribute__((ext_vector_type(8)))  __bf16 v8bf;
typedef __attribute__((ext_vector_type(8)))  float  v8f;

#define S_   512
#define B_   128
#define M_   512
#define N_   512
#define NQ_  2
#define NWG  32     // persistent workgroups
#define NTHR 256    // 8 wave32 per workgroup -> 256 waves total

__device__ __forceinline__ float sigmoidf_(float x) { return 1.0f / (1.0f + __expf(-x)); }

// ---- WMMA fragment loads, per CDNA5 ISA VGPR layouts (CDNA5 ISA) ----------

// A-matrix 16x32 bf16 from an LDS-staged 16x512 row-major tile.
// lanes 0-15: M=lane, K = kk+{0..7} (v0-3) and kk+{16..23} (v4-7)
// lanes16-31: M=lane-16, K = kk+{8..15} and kk+{24..31}
__device__ __forceinline__ v16bf wmma_ldA_lds(const bf16* base, int kk) {
  const int lane = threadIdx.x & 31;
  const int m  = lane & 15;
  const int k0 = kk + ((lane >> 4) << 3);
  const bf16* p = base + m * 512 + k0;
  v8bf lo = *(const v8bf*)(p);
  v8bf hi = *(const v8bf*)(p + 16);
  v16bf a;
#pragma unroll
  for (int e = 0; e < 8; ++e) { a[e] = lo[e]; a[e + 8] = hi[e]; }
  return a;
}

// B-matrix 32x16 bf16. Weight stored "column-major": column j contiguous
// over K with stride ldk. lanes 0-15: N=lane, K=kk+{0..7},kk+{16..23}; lanes16-31: +8.
__device__ __forceinline__ v16bf wmma_ldB(const bf16* __restrict__ base, int ldk,
                                          int col_base, int kk) {
  const int lane = threadIdx.x & 31;
  const int j  = col_base + (lane & 15);
  const int k0 = kk + ((lane >> 4) << 3);
  const bf16* p = base + j * ldk + k0;
  v8bf lo = *(const v8bf*)(p);
  v8bf hi = *(const v8bf*)(p + 16);
  v16bf b;
#pragma unroll
  for (int e = 0; e < 8; ++e) { b[e] = lo[e]; b[e + 8] = hi[e]; }
  return b;
}

__device__ __forceinline__ v8f wmma_bf16(v16bf a, v16bf b, v8f c) {
  // 8 args: (neg_a, A, neg_b, B, c_mod, C, reuse_a, reuse_b)
  return __builtin_amdgcn_wmma_f32_16x16x32_bf16(false, a, false, b, (short)0, c,
                                                 false, false);
}

// Stage 16 activation rows (16x512 bf16 = 16KB) into LDS, vectorized 16B.
__device__ __forceinline__ void stage_rows(bf16* sdst, const bf16* __restrict__ src,
                                           int rowbase) {
#pragma unroll
  for (int i = threadIdx.x * 8; i < 16 * 512; i += NTHR * 8) {
    const int r = i >> 9;
    const int c = i & 511;
    *(v8bf*)(sdst + i) = *(const v8bf*)(src + (rowbase + r) * 512 + c);
  }
}

// ---- grid-wide barrier (L2 atomic counter, monotonic generations) ----------
__device__ __forceinline__ void grid_barrier(unsigned* cnt, unsigned target) {
  __syncthreads();
  __threadfence();                      // release all prior writes to DEV scope
  if (threadIdx.x == 0) {
    atomicAdd(cnt, 1u);
    while (__hip_atomic_load(cnt, __ATOMIC_ACQUIRE, __HIP_MEMORY_SCOPE_AGENT) < target) {
      __builtin_amdgcn_s_sleep(2);
    }
  }
  __syncthreads();
  __threadfence();                      // acquire for all threads
}

// ---- mogrify phase: acc = act[rows,:] @ W^T tile; tgt = 2*sigmoid(acc)*tgt -
// C/D layout: lanes 0-15: N=lane, v r holds M=r; lanes16-31: N=lane-16, M=8+r.
__device__ __forceinline__ void mogrify_phase(bf16* sA,
                                              const bf16* __restrict__ act,
                                              const bf16* __restrict__ W,
                                              float* tgt_f, bf16* tgt_b,
                                              int rowbase, int colbase) {
  stage_rows(sA, act, rowbase);
  __syncthreads();
  v8f acc = {};
#pragma unroll 4
  for (int kk = 0; kk < 512; kk += 32) {
    v16bf b = wmma_ldB(W, 512, colbase, kk);
    if (kk + 32 < 512)
      __builtin_prefetch(W + (colbase + (threadIdx.x & 15)) * 512 + kk + 32, 0, 1);
    v16bf a = wmma_ldA_lds(sA, kk);
    acc = wmma_bf16(a, b, acc);
  }
  const int lane = threadIdx.x & 31;
  const int n  = colbase + (lane & 15);
  const int mb = rowbase + ((lane >> 4) << 3);
#pragma unroll
  for (int r = 0; r < 8; ++r) {
    const int m = mb + r;
    float v = 2.0f * sigmoidf_(acc[r]) * tgt_f[m * 512 + n];
    tgt_f[m * 512 + n] = v;
    tgt_b[m * 512 + n] = (bf16)v;
  }
}

// ---- one-time precompute: Q = Ql@Qr, R = Rl@Rr (row-major, bf16) -----------
__global__ void mog_pre_qr(const float* __restrict__ Ql, const float* __restrict__ Qr,
                           const float* __restrict__ Rl, const float* __restrict__ Rr,
                           bf16* __restrict__ QB, bf16* __restrict__ RB) {
  int idx = blockIdx.x * 256 + threadIdx.x;        // 0 .. 2*512*512-1
  int i   = idx >> 18;                             // pair
  int rem = idx & 262143;
  int m   = rem >> 9;
  int n   = rem & 511;
  const float* ql = Ql + i * 512 * 256 + m * 256;  // [NQ, M, K]
  const float* qr = Qr + i * 256 * 512 + n;        // [NQ, K, N]
  float a = 0.0f;
  for (int k = 0; k < 256; ++k) a += ql[k] * qr[k * 512];
  QB[idx] = (bf16)a;
  const float* rl = Rl + i * 512 * 256 + m * 256;  // [NQ, N, K]
  const float* rr = Rr + i * 256 * 512 + n;        // [NQ, K, M]
  float b = 0.0f;
  for (int k = 0; k < 256; ++k) b += rl[k] * rr[k * 512];
  RB[idx] = (bf16)b;
}

// Wcat[j][k] = k<512 ? W_ih[j,k] : W_hh[j,k-512]  (column j contiguous over K)
__global__ void mog_pre_w(const float* __restrict__ Wih, const float* __restrict__ Whh,
                          const float* __restrict__ bih, const float* __restrict__ bhh,
                          bf16* __restrict__ Wcat, float* __restrict__ bias,
                          unsigned* cnt) {
  long idx = (long)blockIdx.x * 256 + threadIdx.x;  // 0 .. 2048*1024-1
  if (idx == 0) *cnt = 0u;                          // reset barrier each launch
  if (idx < 2048) bias[idx] = bih[idx] + bhh[idx];
  if (idx < 2048L * 1024) {
    int j = (int)(idx >> 10);
    int k = (int)(idx & 1023);
    float v = (k < 512) ? Wih[j * 512 + k] : Whh[j * 512 + (k - 512)];
    Wcat[idx] = (bf16)v;
  }
}

// ---- persistent scan kernel ------------------------------------------------
__global__ void __launch_bounds__(NTHR, 1)
mog_persistent(const float* __restrict__ x, float* __restrict__ out,
               const bf16* __restrict__ QB, const bf16* __restrict__ RB,
               const bf16* __restrict__ Wcat, const float* __restrict__ bias,
               float* xx_f, bf16* xx_b0, bf16* xx_b1,
               float* h_f,  bf16* h_b0,  bf16* h_b1,
               float* c_f, unsigned* cnt) {
  __shared__ bf16 sA[16 * 1024];        // 32KB: [0..8K) xx rows, [8K..16K) h rows

  // block = one row-group (16 rows) x 8 column stripes
  const int rowbase = (blockIdx.x >> 2) * 16;                         // 0..112
  const int colbase = ((blockIdx.x & 3) * 8 + (threadIdx.x >> 5)) * 16; // 0..496
  const int gtid    = blockIdx.x * NTHR + threadIdx.x;
  unsigned  bars    = 0;

  // init: h=c=0, xx = x[0]
  for (int i = gtid; i < B_ * N_; i += NWG * NTHR) {
    h_f[i] = 0.0f; h_b0[i] = (bf16)0.0f; c_f[i] = 0.0f;
    float v = x[i];
    xx_f[i] = v; xx_b0[i] = (bf16)v;
  }
  ++bars; grid_barrier(cnt, bars * NWG);

  for (int t = 0; t < S_; ++t) {
    bf16* xxb_cur = (t & 1) ? xx_b1 : xx_b0;
    bf16* xxb_nxt = (t & 1) ? xx_b0 : xx_b1;
    bf16* hb_cur  = (t & 1) ? h_b1  : h_b0;
    bf16* hb_nxt  = (t & 1) ? h_b0  : h_b1;

    // mogrification: 2 pairs of (x-update, h-update)
#pragma unroll 1
    for (int p = 0; p < NQ_; ++p) {
      mogrify_phase(sA, hb_cur, QB + p * (M_ * N_), xx_f, xxb_cur, rowbase, colbase);
      ++bars; grid_barrier(cnt, bars * NWG);
      mogrify_phase(sA, xxb_cur, RB + p * (N_ * M_), h_f, hb_cur, rowbase, colbase);
      ++bars; grid_barrier(cnt, bars * NWG);
    }

    // gate GEMM: [xx|h] (K=1024) x Wcat columns {n, 512+n, 1024+n, 1536+n}
    stage_rows(sA, xxb_cur, rowbase);
    stage_rows(sA + 16 * 512, hb_cur, rowbase);
    __syncthreads();
    v8f acc[4] = {};
#pragma unroll 2
    for (int kk = 0; kk < 1024; kk += 32) {
      const bf16* a_src = sA + ((kk < 512) ? 0 : 16 * 512);
      v16bf a = wmma_ldA_lds(a_src, kk & 511);
#pragma unroll
      for (int g = 0; g < 4; ++g) {
        v16bf b = wmma_ldB(Wcat, 1024, g * 512 + colbase, kk);
        acc[g] = wmma_bf16(a, b, acc[g]);
      }
    }

    // LSTM cell update, in-register
    const int lane = threadIdx.x & 31;
    const int n   = colbase + (lane & 15);
    const float bi  = bias[n];
    const float bf2 = bias[512 + n];
    const float bg  = bias[1024 + n];
    const float bo  = bias[1536 + n];
    const int mb = rowbase + ((lane >> 4) << 3);
    float* outt = out + (size_t)t * (B_ * N_);
#pragma unroll
    for (int r = 0; r < 8; ++r) {
      const int m = mb + r;
      float ig = sigmoidf_(acc[0][r] + bi);
      float fg = sigmoidf_(acc[1][r] + bf2);
      float gg = tanhf(acc[2][r] + bg);
      float og = sigmoidf_(acc[3][r] + bo);
      float cv = fg * c_f[m * 512 + n] + ig * gg;
      float hv = og * tanhf(cv);
      c_f[m * 512 + n] = cv;
      h_f[m * 512 + n] = hv;
      hb_nxt[m * 512 + n] = (bf16)hv;   // double-buffered: no race w/ readers
      outt[m * 512 + n] = hv;
    }

    // stage next step's xx (fused into this phase; separate buffer -> no race)
    if (t + 1 < S_) {
      const float* xt = x + (size_t)(t + 1) * (B_ * M_);
      for (int i = gtid; i < B_ * M_; i += NWG * NTHR) {
        float v = xt[i];
        xx_f[i] = v; xxb_nxt[i] = (bf16)v;
      }
    }
    ++bars; grid_barrier(cnt, bars * NWG);
  }

  // tuple tail: final h then c
  const size_t base = (size_t)S_ * B_ * N_;
  for (int i = gtid; i < B_ * N_; i += NWG * NTHR) {
    out[base + i] = h_f[i];
    out[base + B_ * N_ + i] = c_f[i];
  }
}

// ---------------------------------------------------------------------------
extern "C" void kernel_launch(void* const* d_in, const int* in_sizes, int n_in,
                              void* d_out, int out_size, void* d_ws, size_t ws_size,
                              hipStream_t stream) {
  (void)in_sizes; (void)n_in; (void)out_size; (void)ws_size;
  const float* x   = (const float*)d_in[0];
  const float* Ql  = (const float*)d_in[1];
  const float* Qr  = (const float*)d_in[2];
  const float* Rl  = (const float*)d_in[3];
  const float* Rr  = (const float*)d_in[4];
  const float* Wih = (const float*)d_in[5];
  const float* Whh = (const float*)d_in[6];
  const float* bih = (const float*)d_in[7];
  const float* bhh = (const float*)d_in[8];
  float* out = (float*)d_out;

  // workspace carve-up (~7.6 MB)
  char* ws = (char*)d_ws;
  size_t off = 0;
  auto take = [&](size_t bytes) -> void* {
    void* p = ws + off;
    off = (off + bytes + 255) & ~(size_t)255;
    return p;
  };
  unsigned* cnt   = (unsigned*)take(256);
  bf16*  QB    = (bf16*)take((size_t)NQ_ * M_ * N_ * sizeof(bf16));
  bf16*  RB    = (bf16*)take((size_t)NQ_ * N_ * M_ * sizeof(bf16));
  bf16*  Wcat  = (bf16*)take((size_t)2048 * 1024 * sizeof(bf16));
  float* bias  = (float*)take(2048 * sizeof(float));
  float* xx_f  = (float*)take((size_t)B_ * M_ * sizeof(float));
  bf16*  xx_b0 = (bf16*)take((size_t)B_ * M_ * sizeof(bf16));
  bf16*  xx_b1 = (bf16*)take((size_t)B_ * M_ * sizeof(bf16));
  float* h_f   = (float*)take((size_t)B_ * N_ * sizeof(float));
  bf16*  h_b0  = (bf16*)take((size_t)B_ * N_ * sizeof(bf16));
  bf16*  h_b1  = (bf16*)take((size_t)B_ * N_ * sizeof(bf16));
  float* c_f   = (float*)take((size_t)B_ * N_ * sizeof(float));

  // one-time precompute (re-run every call: deterministic, graph-safe)
  mog_pre_qr<<<dim3(2048), dim3(256), 0, stream>>>(Ql, Qr, Rl, Rr, QB, RB);
  mog_pre_w<<<dim3(8192), dim3(256), 0, stream>>>(Wih, Whh, bih, bhh, Wcat, bias, cnt);

  // persistent scan
  mog_persistent<<<dim3(NWG), dim3(NTHR), 0, stream>>>(
      x, out, QB, RB, Wcat, bias, xx_f, xx_b0, xx_b1, h_f, h_b0, h_b1, c_f, cnt);
}
